// Composer_31413390803006
// MI455X (gfx1250) — matrix operand order repaired and checked by reading, not compiled, not run on hardware
//
#include <hip/hip_runtime.h>
#include <hip/hip_bf16.h>
#include <math.h>

// Problem constants (from reference)
#define Bb 8
#define Ss 512
#define Dd 384
#define BS (Bb*Ss)            // 4096 rows
#define KTOT (Dd*Dd)          // 147456 reduction length
#define KSPLIT 48
#define KCHUNK (KTOT/KSPLIT)  // 3072 = 8 d-values * 384

typedef __attribute__((ext_vector_type(16))) __bf16 v16bf;
typedef __attribute__((ext_vector_type(8)))  float  v8f;
typedef __attribute__((ext_vector_type(4)))  unsigned int v4u;
typedef __attribute__((ext_vector_type(8)))  int v8i;
typedef __attribute__((ext_vector_type(4)))  int v4i;

static __device__ __forceinline__ __bf16 f2bf(float f) { return (__bf16)f; }

// ---------------------------------------------------------------------------
// 1) Gather token rows, compute dep=tanh(tok), store both as bf16, zero u,
//    and compute the pruning flag: row (b,s2) matters iff heads[b,heads[b,s2]]==0
// ---------------------------------------------------------------------------
__global__ void prep_kernel(const int* __restrict__ tokens,
                            const int* __restrict__ heads,
                            const float* __restrict__ tok_table,
                            __bf16* __restrict__ tok_bf,
                            __bf16* __restrict__ dep_bf,
                            float* __restrict__ u,
                            int* __restrict__ flags) {
  const int row = blockIdx.x;
  const int tid = threadIdx.x;
  const int tok = tokens[row];
  const float* src = tok_table + (size_t)tok * Dd;
  for (int o = tid; o < Dd; o += blockDim.x) {
    float tv = src[o];
    tok_bf[(size_t)row * Dd + o] = f2bf(tv);
    dep_bf[(size_t)row * Dd + o] = f2bf(tanhf(tv));
    u[(size_t)row * Dd + o] = 0.0f;
  }
  if (tid == 0) {
    int b = row / Ss;
    int h = heads[row];
    flags[row] = (heads[b * Ss + h] == 0) ? 1 : 0;
  }
}

// ---------------------------------------------------------------------------
// 2) Deterministic compaction: exclusive scan over 4096 flags (one block).
//    Produces row_ids in ascending order (required by TDM gather-mode OOB).
// ---------------------------------------------------------------------------
__global__ void scan_kernel(const int* __restrict__ flags,
                            int* __restrict__ row_ids,
                            int* __restrict__ count) {
  __shared__ int part[1024];
  const int t = threadIdx.x;
  int local[4];
  int s = 0;
  #pragma unroll
  for (int j = 0; j < 4; ++j) { local[j] = flags[t * 4 + j]; s += local[j]; }
  part[t] = s;
  __syncthreads();
  for (int off = 1; off < 1024; off <<= 1) {
    int v = (t >= off) ? part[t - off] : 0;
    __syncthreads();
    part[t] += v;
    __syncthreads();
  }
  int run = part[t] - s;            // exclusive prefix
  #pragma unroll
  for (int j = 0; j < 4; ++j) {
    if (local[j]) { row_ids[run] = t * 4 + j; ++run; }
  }
  if (t == 1023) *count = part[1023];
}

// ---------------------------------------------------------------------------
// 3) Constants: sum(Wr), base[o] = tanh(bc[o])*sum(Wr) + br
// ---------------------------------------------------------------------------
__global__ void const_kernel(const float* __restrict__ Wr,
                             const float* __restrict__ br,
                             const float* __restrict__ bc,
                             float* __restrict__ basev) {
  __shared__ float red[512];
  const int t = threadIdx.x;
  red[t] = Wr[t];
  __syncthreads();
  for (int off = 256; off > 0; off >>= 1) {
    if (t < off) red[t] += red[t + off];
    __syncthreads();
  }
  float sw = red[0];
  if (t < Dd) basev[t] = tanhf(bc[t]) * sw + br[0];
}

// ---------------------------------------------------------------------------
// 4) Output base: rows with heads==0 get base vector, everything else zero
// ---------------------------------------------------------------------------
__global__ void out_init_kernel(const int* __restrict__ heads,
                                const float* __restrict__ basev,
                                float* __restrict__ out) {
  const int row = blockIdx.x;
  const bool on = (heads[row] == 0);
  for (int o = threadIdx.x; o < Dd; o += blockDim.x)
    out[(size_t)row * Dd + o] = on ? basev[o] : 0.0f;
}

// ---------------------------------------------------------------------------
// TDM gather-mode descriptor issue: load up to 16 indirectly-indexed bf16 rows
// of width 384 from `gbase` into LDS at `lds_byte_addr`.
//   group0: count=1 | gather_mode, lds_addr, 57-bit global addr, type=2
//   group1: data_size=2B, tensor_dim0=384, stride0=384, tile_dim0=384,
//           tile_dim1=#valid rows
//   group2/3: sixteen 16-bit row indices (ascending)
// ---------------------------------------------------------------------------
static __device__ __forceinline__ void tdm_gather_rows(
    const __bf16* gbase, unsigned lds_byte_addr,
    const unsigned* idx16, int mvalid) {
  unsigned long long ga = (unsigned long long)(uintptr_t)gbase;

  v4u g0;
  g0[0] = 0x80000001u;                               // count=1, gather_mode=1
  g0[1] = lds_byte_addr;                             // LDS byte address
  g0[2] = (unsigned)(ga & 0xFFFFFFFFu);              // global addr [31:0]
  g0[3] = (unsigned)((ga >> 32) & 0x01FFFFFFu)       // global addr [56:32]
        | 0x80000000u;                               // type = 2 ("image")

  v8i g1;
  g1[0] = (int)(1u << 16);                           // data_size = 1 (2 bytes)
  g1[1] = (int)(((unsigned)Dd & 0xFFFFu) << 16);     // tensor_dim0[15:0]=384
  g1[2] = (int)(((unsigned)BS & 0xFFFFu) << 16);     // tensor_dim1[15:0]=4096
  g1[3] = (int)(((unsigned)Dd & 0xFFFFu) << 16);     // tile_dim0 = 384
  g1[4] = (int)((unsigned)mvalid & 0xFFFFu);         // tile_dim1 = #indices
  g1[5] = (int)(unsigned)Dd;                         // tensor_dim0_stride = 384
  g1[6] = 0;
  g1[7] = 0;

  v4i g2, g3;
  #pragma unroll
  for (int q = 0; q < 4; ++q) {
    g2[q] = (int)(idx16[2 * q]     | (idx16[2 * q + 1] << 16));
    g3[q] = (int)(idx16[8 + 2 * q] | (idx16[8 + 2 * q + 1] << 16));
  }

#if defined(__clang_major__) && (__clang_major__ >= 23)
  v8i gpad = {};
  __builtin_amdgcn_tensor_load_to_lds(g0, g1, g2, g3, gpad, 0);
#else
  __builtin_amdgcn_tensor_load_to_lds(g0, g1, g2, g3, 0);
#endif
}

// ---------------------------------------------------------------------------
// 5) Split-K WMMA GEMM over compacted rows.
//    u[m,o] += sum_k Wc[o,k] * (tok[m,d(k)] * dep[m,e(k)]),  k = d*384+e
//    A (16x32 bf16) = Wc tile (fp32 stream, cvt in flight).
//    B (32x16 bf16) = activation outer product from TDM-gathered LDS rows.
//    Grid: x = 3 N-blocks of 128 o's (8 waves x 16), y = M tiles, z = K split.
// ---------------------------------------------------------------------------
__global__ __launch_bounds__(256)
void bilinear_gemm_kernel(const float* __restrict__ Wc,
                          const __bf16* __restrict__ tok_bf,
                          const __bf16* __restrict__ dep_bf,
                          const int* __restrict__ row_ids,
                          const int* __restrict__ count_p,
                          float* __restrict__ u) {
  const int cnt = *count_p;
  const int m0 = blockIdx.y * 16;
  if (m0 >= cnt) return;                       // expected case: only y==0 live

  __shared__ __bf16 tokS[16][Dd];
  __shared__ __bf16 depS[16][Dd];

  // ---- Stage the M-tile's activation rows via the Tensor Data Mover ------
  // Wave 0 issues two gather-mode TENSOR_LOAD_TO_LDS ops (tok rows, dep rows)
  // and waits on TENSORcnt; the workgroup barrier publishes the LDS data.
  // Unstaged tile rows (m >= mvalid) only pollute their own D columns, which
  // are masked at the store below.
  if (threadIdx.x < 32) {
    const int mvalid = (cnt - m0 < 16) ? (cnt - m0) : 16;
    unsigned idx16[16];
    #pragma unroll
    for (int j = 0; j < 16; ++j) {
      int slot = m0 + j;
      idx16[j] = (slot < cnt) ? (unsigned)row_ids[slot] : 0u;
    }
    tdm_gather_rows(tok_bf, (unsigned)(uintptr_t)&tokS[0][0], idx16, mvalid);
    tdm_gather_rows(dep_bf, (unsigned)(uintptr_t)&depS[0][0], idx16, mvalid);
    __builtin_amdgcn_s_wait_tensorcnt(0);
  }
  __syncthreads();

  const int wave = threadIdx.x >> 5;
  const int lane = threadIdx.x & 31;
  const int half = lane >> 4;                  // K half-split per 16-bit layout
  const int l15  = lane & 15;                  // A row (o) / B col (m) / D col
  const int obase = blockIdx.x * 128 + wave * 16;
  const int o    = obase + l15;
  const int kbase = blockIdx.z * KCHUNK;

  const float* wp = Wc + (size_t)o * KTOT + kbase;
  const int b1 = half * 16;                    // A k_locals {0..7} or {16..23}
  const int b2 = half * 16 + 8;                // A k_locals {8..15} or {24..31}

  v8f acc = {};
  for (int t = 0; t < KCHUNK / 32; ++t) {
    const int klocal = t * 32;
    const int dcur = (kbase + klocal) / Dd;    // 384 % 32 == 0 -> d fixed/chunk
    const int e0   = (kbase + klocal) % Dd;

    const float* ap = wp + klocal;
    __builtin_prefetch(ap + 512, 0, 0);        // stream Wc ahead (global_prefetch)

    float4 a0 = *(const float4*)(ap + b1);
    float4 a1 = *(const float4*)(ap + b1 + 4);
    float4 a2 = *(const float4*)(ap + b2);
    float4 a3 = *(const float4*)(ap + b2 + 4);
    v16bf Amat;
    Amat[0]  = f2bf(a0.x); Amat[1]  = f2bf(a0.y); Amat[2]  = f2bf(a0.z); Amat[3]  = f2bf(a0.w);
    Amat[4]  = f2bf(a1.x); Amat[5]  = f2bf(a1.y); Amat[6]  = f2bf(a1.z); Amat[7]  = f2bf(a1.w);
    Amat[8]  = f2bf(a2.x); Amat[9]  = f2bf(a2.y); Amat[10] = f2bf(a2.z); Amat[11] = f2bf(a2.w);
    Amat[12] = f2bf(a3.x); Amat[13] = f2bf(a3.y); Amat[14] = f2bf(a3.z); Amat[15] = f2bf(a3.w);

    // B[k, m]: lane = column m; this lane covers k = e0 + half*16 + j
    const float tv = (float)tokS[l15][dcur];
    const __bf16* dp = &depS[l15][e0 + half * 16];
    v16bf Bmat;
    #pragma unroll
    for (int j = 0; j < 16; ++j) Bmat[j] = f2bf(tv * (float)dp[j]);

    acc = __builtin_amdgcn_wmma_f32_16x16x32_bf16(false, Amat, false, Bmat,
                                                  (short)0, acc, false, false);
  }

  // D layout: lane l15 -> column (m), VGPR r -> row o = obase + half*8 + r
  const int mslot = m0 + l15;
  if (mslot < cnt) {
    float* up = u + (size_t)mslot * Dd + obase + half * 8;
    #pragma unroll
    for (int r = 0; r < 8; ++r)
      __hip_atomic_fetch_add(up + r, acc[r], __ATOMIC_RELAXED,
                             __HIP_MEMORY_SCOPE_AGENT);
  }
}

// ---------------------------------------------------------------------------
// 6) Epilogue: contrib = (tanh(u+bc) - tanh(bc)) * Wr[s2], scatter to (b, h)
// ---------------------------------------------------------------------------
__global__ void scatter_kernel(const int* __restrict__ row_ids,
                               const int* __restrict__ count_p,
                               const int* __restrict__ heads,
                               const float* __restrict__ u,
                               const float* __restrict__ bc,
                               const float* __restrict__ Wr,
                               float* __restrict__ out) {
  const int slot = blockIdx.x;
  if (slot >= *count_p) return;
  const int r = row_ids[slot];
  const int b = r / Ss, s2 = r % Ss;
  const int h = heads[r];
  const float w = Wr[s2];
  float* op = out + ((size_t)b * Ss + h) * Dd;
  const float* up = u + (size_t)slot * Dd;
  for (int o = threadIdx.x; o < Dd; o += blockDim.x) {
    float c = (tanhf(up[o] + bc[o]) - tanhf(bc[o])) * w;
    __hip_atomic_fetch_add(op + o, c, __ATOMIC_RELAXED, __HIP_MEMORY_SCOPE_AGENT);
  }
}

// ---------------------------------------------------------------------------
extern "C" void kernel_launch(void* const* d_in, const int* in_sizes, int n_in,
                              void* d_out, int out_size, void* d_ws, size_t ws_size,
                              hipStream_t stream) {
  (void)in_sizes; (void)n_in; (void)out_size; (void)ws_size;
  const int*   tokens    = (const int*)d_in[0];
  // d_in[1] = dep_types: unused (the reference discards its embedding)
  const int*   heads     = (const int*)d_in[2];
  const float* tok_table = (const float*)d_in[3];
  const float* Wc        = (const float*)d_in[4];
  const float* bc        = (const float*)d_in[5];
  const float* Wr        = (const float*)d_in[6];
  const float* br        = (const float*)d_in[7];
  float* out = (float*)d_out;

  char* ws = (char*)d_ws;
  size_t off = 0;
  auto carve = [&](size_t bytes) -> char* {
    char* p = ws + off;
    off = (off + bytes + 255) & ~(size_t)255;
    return p;
  };
  __bf16* tok_bf  = (__bf16*)carve((size_t)BS * Dd * sizeof(__bf16)); // 3.0 MB
  __bf16* dep_bf  = (__bf16*)carve((size_t)BS * Dd * sizeof(__bf16)); // 3.0 MB
  float*  u       = (float*) carve((size_t)BS * Dd * sizeof(float));  // 6.3 MB
  float*  basev   = (float*) carve(Dd * sizeof(float));
  int*    flags   = (int*)   carve(BS * sizeof(int));
  int*    row_ids = (int*)   carve(BS * sizeof(int));
  int*    count   = (int*)   carve(256);

  prep_kernel<<<BS, 128, 0, stream>>>(tokens, heads, tok_table,
                                      tok_bf, dep_bf, u, flags);
  scan_kernel<<<1, 1024, 0, stream>>>(flags, row_ids, count);
  const_kernel<<<1, 512, 0, stream>>>(Wr, br, bc, basev);
  out_init_kernel<<<BS, 128, 0, stream>>>(heads, basev, out);
  bilinear_gemm_kernel<<<dim3(3, BS / 16, KSPLIT), 256, 0, stream>>>(
      Wc, tok_bf, dep_bf, row_ids, count, u);
  scatter_kernel<<<BS, 128, 0, stream>>>(row_ids, count, heads, u, bc, Wr, out);
}
